// GraphAttentionPooling_82729660056046
// MI455X (gfx1250) — matrix operand order repaired and checked by reading, not compiled, not run on hardware
//
#include <hip/hip_runtime.h>
#include <hip/hip_bf16.h>

// Problem constants (fixed by the reference).
constexpr int Nn = 1 << 20;   // rows
constexpr int Hh = 128;       // features
constexpr int Bb = 128;       // segments
constexpr float NEG_SLOPE = 0.01f;

typedef __attribute__((ext_vector_type(16))) _Float16 v16h;
typedef __attribute__((ext_vector_type(8)))  float    v8f;

union AFrag { v16h v; _Float16 e[16]; };

// Monotonic float<->uint encoding for atomic max on floats.
__device__ __forceinline__ unsigned fenc(float x) {
    unsigned u = __float_as_uint(x);
    return (u & 0x80000000u) ? ~u : (u | 0x80000000u);
}
__device__ __forceinline__ float fdec(unsigned e) {
    unsigned u = (e & 0x80000000u) ? (e & 0x7FFFFFFFu) : ~e;
    return __uint_as_float(u);
}

// ---------------------------------------------------------------------------
// Kernel 1: q_score[b] = Aq[b,:] . w_q ; zero out/m/denom (we accumulate with
// atomics later, and the harness does not re-zero between replays).
// ---------------------------------------------------------------------------
__global__ void k_init(const float* __restrict__ aq, const float* __restrict__ w,
                       float* __restrict__ qs, unsigned* __restrict__ m_u,
                       float* __restrict__ denom, float* __restrict__ out) {
    int b = threadIdx.x;            // 128 threads
    const float* wq = w + Hh;
    float s = 0.f;
    #pragma unroll 8
    for (int k = 0; k < Hh; ++k) s += aq[b * Hh + k] * wq[k];
    qs[b] = s;
    m_u[b] = 0u;                    // decodes to NaN -> treated as "empty segment"
    denom[b] = 0.f;
    for (int k = 0; k < Hh; ++k) out[b * Hh + k] = 0.f;
}

// ---------------------------------------------------------------------------
// Kernel 2: pass 1 over h. One wave per row (wave32): float4 per lane =
// coalesced 512B row load; xor-shuffle reduction; leaky ReLU; store gate;
// segment max via LDS atomicMax then one global atomicMax per touched segment.
// Block = 256 threads = 8 waves, 8 rows per wave -> 64 rows/block.
// ---------------------------------------------------------------------------
__global__ void k_gate(const float* __restrict__ h, const int* __restrict__ seg,
                       const float* __restrict__ w, const float* __restrict__ qs,
                       float* __restrict__ gate, unsigned* __restrict__ m_u) {
    __shared__ unsigned bmax[Bb];
    int t = threadIdx.x;
    for (int i = t; i < Bb; i += 256) bmax[i] = 0u;
    __syncthreads();

    int lane = t & 31, wv = t >> 5;
    const float4 w4 = ((const float4*)w)[lane];      // w_h[lane*4 .. +3]
    int base = blockIdx.x * 64 + wv * 8;

    for (int rr = 0; rr < 8; ++rr) {
        int row = base + rr;
        float4 hv = ((const float4*)(h + (size_t)row * Hh))[lane];
        float p = hv.x * w4.x + hv.y * w4.y + hv.z * w4.z + hv.w * w4.w;
        #pragma unroll
        for (int off = 16; off > 0; off >>= 1) p += __shfl_xor(p, off, 32);
        if (lane == 0) {
            int sg = seg[row];
            float g = p + qs[sg];
            g = (g >= 0.f) ? g : NEG_SLOPE * g;
            gate[row] = g;
            atomicMax(&bmax[sg], fenc(g));
        }
    }
    __syncthreads();
    for (int i = t; i < Bb; i += 256)
        if (bmax[i]) atomicMax(&m_u[i], bmax[i]);
}

// ---------------------------------------------------------------------------
// Kernel 3: finalize segment max: decode, non-finite -> 0 (matches reference).
// ---------------------------------------------------------------------------
__global__ void k_finm(const unsigned* __restrict__ m_u, float* __restrict__ m) {
    int b = threadIdx.x;
    float f = fdec(m_u[b]);
    if (!__builtin_isfinite(f)) f = 0.f;
    m[b] = f;
}

// ---------------------------------------------------------------------------
// Kernel 4: e = exp(gate - m[seg]) in place; denom[b] += segment partials
// (LDS partial sums cut global atomic traffic by 256x).
// ---------------------------------------------------------------------------
__global__ void k_exp(float* __restrict__ ebuf, const int* __restrict__ seg,
                      const float* __restrict__ m, float* __restrict__ denom) {
    __shared__ float part[Bb];
    int t = threadIdx.x;
    for (int i = t; i < Bb; i += 256) part[i] = 0.f;
    __syncthreads();
    int idx = blockIdx.x * 256 + t;
    int sg = seg[idx];
    float e = __expf(ebuf[idx] - m[sg]);
    ebuf[idx] = e;
    atomicAdd(&part[sg], e);
    __syncthreads();
    for (int i = t; i < Bb; i += 256)
        if (part[i] != 0.f) atomicAdd(&denom[i], part[i]);
}

// ---------------------------------------------------------------------------
// Kernel 5: pass 2 over h. pooled[b,:] += alpha[n] * h[n,:] via WMMA, with
// CDNA5 async memory->LDS staging (GLOBAL_LOAD_ASYNC_TO_LDS_B128, ASYNCcnt)
// and double buffering: issue tile t+1's async loads, compute tile t.
// Block = 256 threads = 8 waves; wave w owns feature block [w*16, w*16+16).
// Block covers 512 rows in 16 tiles of 32 rows. Per tile:
//   B(32x16) f16  = h tile (from LDS)
//   A(16x32) f16  = alpha gated by segment one-hot relative to sbase
//   C(16x16) f32 += A x B   (v_wmma_f32_16x16x32_f16)
// seg ids are sorted, so sbase (16-segment window) rarely advances; on
// advance we flush C with float atomicAdd.
// ---------------------------------------------------------------------------
__global__ void k_pool(const float* __restrict__ h, const int* __restrict__ seg,
                       const float* __restrict__ ebuf, const float* __restrict__ denom,
                       float* __restrict__ out) {
    __shared__ float s_h[2][32 * Hh];   // 2 x 16 KB tiles (double buffer)
    __shared__ float s_alpha[2][32];
    __shared__ int   s_seg[2][32];

    const int t    = threadIdx.x;
    const int lane = t & 31;
    const int fb   = t >> 5;                  // wave id = feature block 0..7
    const int r0   = blockIdx.x * 512;

    const int Mrow   = lane & 15;             // A-matrix row owned by this lane
    const int koffA  = (lane >= 16) ? 8 : 0;  // A layout: lanes 16-31 hold K+8
    const int kbaseB = (lane >= 16) ? 8 : 0;  // B layout: lanes 16-31 hold K+8
    const int colB   = fb * 16 + (lane & 15); // feature column this lane feeds

    v8f cacc = {};
    int sbase = 0;
    bool first = true;
    AFrag au, bu;

    // Issue async global->LDS staging for one 32x128 f32 tile (16 KB):
    // 1024 b128 transfers, 4 per thread, tracked by ASYNCcnt (no VGPR staging).
    auto stage = [&](int tile, int buf) {
        unsigned ldsbase = (unsigned)(size_t)(&s_h[buf][0]);   // LDS offset = addr[31:0]
        unsigned gbase   = (unsigned)((size_t)(r0 + tile * 32) * Hh * sizeof(float));
        #pragma unroll
        for (int j = 0; j < 4; ++j) {
            unsigned ldsa = ldsbase + (unsigned)(t + 256 * j) * 16u;
            unsigned goff = gbase   + (unsigned)(t + 256 * j) * 16u;
            asm volatile("global_load_async_to_lds_b128 %0, %1, %2"
                         :: "v"(ldsa), "v"(goff), "s"(h) : "memory");
        }
        if (t < 32) {
            int row = r0 + tile * 32 + t;
            int sg = seg[row];
            float ev = ebuf[row];
            float dn = denom[sg];
            s_alpha[buf][t] = ev / fmaxf(dn, 1e-20f);
            s_seg[buf][t] = sg;
        }
    };

    auto flushC = [&]() {
        #pragma unroll
        for (int j = 0; j < 8; ++j) {
            int sgi = sbase + j + ((lane >> 4) << 3);   // M = j (+8 for hi lanes)
            if (sgi < Bb) atomicAdd(&out[sgi * Hh + colB], cacc[j]);
        }
        cacc = (v8f){};
    };

    stage(0, 0);

    for (int t16 = 0; t16 < 16; ++t16) {
        const int buf = t16 & 1;
        // Wait for this wave's async loads, then make all waves' LDS visible.
        asm volatile("s_wait_asynccnt 0" ::: "memory");
        __syncthreads();
        // Kick off next tile into the other buffer; overlaps with compute below.
        if (t16 + 1 < 16) stage(t16 + 1, buf ^ 1);

        auto buildA = [&]() {
            #pragma unroll
            for (int i = 0; i < 16; ++i) {
                int K = ((i >= 8) ? 16 : 0) + koffA + (i & 7);
                int sg = s_seg[buf][K];
                float a = s_alpha[buf][K];
                au.e[i] = (sg - sbase == Mrow) ? (_Float16)a : (_Float16)0.f;
            }
        };

        // Build B fragment from LDS (lanes 0-15 read contiguous 64B rows).
        #pragma unroll
        for (int i = 0; i < 16; ++i)
            bu.e[i] = (_Float16)s_h[buf][(((i >= 8) ? 16 : 0) + kbaseB + (i & 7)) * Hh + colB];

        int tmin = __builtin_amdgcn_readfirstlane(s_seg[buf][0]);
        int tmax = __builtin_amdgcn_readfirstlane(s_seg[buf][31]);
        if (first) { sbase = tmin; first = false; }
        if (tmin >= sbase + 16) { flushC(); sbase = tmin; }

        buildA();
        cacc = __builtin_amdgcn_wmma_f32_16x16x32_f16(
            false, au.v, false, bu.v, (short)0, cacc, false, false);

        // Rare: tile spans beyond current 16-segment window.
        while (tmax >= sbase + 16) {
            flushC();
            int ns = sbase + 16, nb = tmax;
            for (int k = 0; k < 32; ++k) {
                int sk = __builtin_amdgcn_readfirstlane(s_seg[buf][k]);
                if (sk >= ns) { nb = sk; break; }
            }
            sbase = nb;
            buildA();
            cacc = __builtin_amdgcn_wmma_f32_16x16x32_f16(
                false, au.v, false, bu.v, (short)0, cacc, false, false);
        }
    }
    flushC();
}

// ---------------------------------------------------------------------------
extern "C" void kernel_launch(void* const* d_in, const int* in_sizes, int n_in,
                              void* d_out, int out_size, void* d_ws, size_t ws_size,
                              hipStream_t stream) {
    const float* h   = (const float*)d_in[0];
    const float* aq  = (const float*)d_in[1];
    const float* w   = (const float*)d_in[2];
    const int*   seg = (const int*)d_in[3];
    float* out = (float*)d_out;

    char* ws = (char*)d_ws;
    float*    ebuf  = (float*)ws;                                  // N floats (gate, then e)
    size_t    off   = (size_t)Nn * sizeof(float);
    float*    qs    = (float*)(ws + off);             off += 512;
    unsigned* m_u   = (unsigned*)(ws + off);          off += 512;
    float*    m     = (float*)(ws + off);             off += 512;
    float*    denom = (float*)(ws + off);

    k_init<<<1, 128, 0, stream>>>(aq, w, qs, m_u, denom, out);
    k_gate<<<Nn / 64, 256, 0, stream>>>(h, seg, w, qs, ebuf, m_u);
    k_finm<<<1, 128, 0, stream>>>(m_u, m);
    k_exp<<<Nn / 256, 256, 0, stream>>>(ebuf, seg, m, denom);
    k_pool<<<Nn / 512, 256, 0, stream>>>(h, seg, ebuf, denom, out);
}
